// ADM_Softmax_32925219291429
// MI455X (gfx1250) — matrix operand order repaired and checked by reading, not compiled, not run on hardware
//
#include <hip/hip_runtime.h>

// Raw ext-vector types (trivial, safe in unions; HIP's uint4/float2 are classes)
typedef __attribute__((ext_vector_type(16))) _Float16     v16h;
typedef __attribute__((ext_vector_type(8)))  float        v8f;
typedef __attribute__((ext_vector_type(4)))  unsigned int u32x4;

#define BM 128
#define BN 128
#define BK 32
#define LDSK 40            // padded stride in halves (80B): conflict-avoiding, 16B-aligned frags

#define MARGIN_R 0.4f
#define MARGIN_F 0.1f
#define SCALE_   5.0f
#define EPS_     1e-12f

union FragU { u32x4 u[2]; v16h h; };

__global__ __launch_bounds__(256)
void adm_softmax_wmma_kernel(const float* __restrict__ feats,
                             const int*   __restrict__ labels,
                             const float* __restrict__ weight,
                             float* __restrict__ out,
                             int Bn, int Dn, int Cn)
{
    // Double-buffered operand tiles: 1 barrier per k-step, loads overlap WMMA.
    __shared__ __align__(16) _Float16 aLds[2][BM * LDSK];  // feats tile  [m][k] f16
    __shared__ __align__(16) _Float16 bLds[2][BN * LDSK];  // weight tile [c][k] f16 (transposed)
    __shared__ float rowpart[BM * 16];                     // per-row sumsq partials
    __shared__ float colpart[2 * BN];                      // per-col sumsq partials
    __shared__ float rfArr[BM];                            // 1/||feats_row||
    __shared__ float rwArr[BN];                            // 1/||W_col||
    __shared__ int   labArr[BM];
    __shared__ float margArr[BM];

    const int t    = threadIdx.x;
    const int lane = t & 31;
    const int wid  = t >> 5;       // 0..7
    const int wm   = wid >> 2;     // wave row 0..1 (64 rows each)
    const int wn   = wid & 3;      // wave col 0..3 (32 cols each)
    const int g    = lane >> 4;    // half-wave group per ISA layout
    const int ln16 = lane & 15;

    const int m0 = blockIdx.x * BM;   // M on grid.x -> blocks sharing a W tile adjacent (L2 reuse)
    const int c0 = blockIdx.y * BN;

    // Staging coordinates. OOB rows/cols are CLAMPED, not masked: a duplicated
    // row/column only feeds accumulators and sum-squares whose outputs are
    // never stored, so no exec-mask branches are needed in the hot loop.
    const int a_row = t >> 4;                   // 0..15
    const int a_k   = 2 * (t & 15);             // even k
    const int b_c   = t & 127;
    const int b_k0  = 2 * (t >> 7);             // 0 or 2
    const int b_cgc = min(c0 + b_c, Cn - 1);    // clamped column

    v8f acc[4][2];
    {
        v8f z = {};
        for (int i = 0; i < 4; ++i)
            for (int j = 0; j < 2; ++j) acc[i][j] = z;
    }
    float asum[8];
    for (int j = 0; j < 8; ++j) asum[j] = 0.f;
    float bsum = 0.f;

    float ra[16];   // register-staged feats  (8 x float2)
    float rb[16];   // register-staged weight (8 x row-pair)

    // Preload labels / margins for this row tile (consumed only after barriers)
    if (t < BM) {
        int mg  = m0 + t;
        int lab = (mg < Bn) ? labels[mg] : -1;
        labArr[t]  = lab;
        margArr[t] = (lab == 0) ? MARGIN_R : MARGIN_F;
    }

    auto loadRegs = [&](int k0) {
        #pragma unroll
        for (int j = 0; j < 8; ++j) {
            int mg = min(m0 + a_row + 16 * j, Bn - 1);           // clamp, no branch
            const float* p = feats + (size_t)mg * Dn + (k0 + a_k);
            ra[2 * j]     = p[0];
            ra[2 * j + 1] = p[1];
        }
        #pragma unroll
        for (int i = 0; i < 8; ++i) {
            int k = b_k0 + 4 * i;
            const float* p = weight + (size_t)(k0 + k) * Cn + b_cgc;
            rb[2 * i]     = __builtin_nontemporal_load(p);       // streamed once
            rb[2 * i + 1] = __builtin_nontemporal_load(p + Cn);
        }
    };

    auto storeStage = [&](int buf) {
        _Float16* ab = aLds[buf];
        _Float16* bb = bLds[buf];
        #pragma unroll
        for (int j = 0; j < 8; ++j) {
            float x0 = ra[2 * j], x1 = ra[2 * j + 1];
            asum[j] += x0 * x0 + x1 * x1;                        // fused row sumsq
            union { _Float16 h[2]; unsigned int u; } pk;
            pk.h[0] = (_Float16)x0; pk.h[1] = (_Float16)x1;
            *(unsigned int*)(ab + (a_row + 16 * j) * LDSK + a_k) = pk.u;
        }
        #pragma unroll
        for (int i = 0; i < 8; ++i) {
            float w0 = rb[2 * i], w1 = rb[2 * i + 1];
            bsum += w0 * w0 + w1 * w1;                           // fused col sumsq
            union { _Float16 h[2]; unsigned int u; } pk;
            pk.h[0] = (_Float16)w0; pk.h[1] = (_Float16)w1;
            *(unsigned int*)(bb + b_c * LDSK + (b_k0 + 4 * i)) = pk.u;
        }
    };

    const int nk = Dn / BK;        // K=512 -> 16 steps

    loadRegs(0);
    storeStage(0);

    for (int ks = 0; ks < nk; ++ks) {
        __syncthreads();                             // single barrier per k-step
        const bool more = (ks + 1 < nk);             // uniform branch
        if (more) loadRegs((ks + 1) * BK);           // next-stage HBM loads overlap WMMA below

        const _Float16* ab = aLds[ks & 1];
        const _Float16* bb = bLds[ks & 1];

        // Fragment loads: ds_load_b128 pairs matching ISA 16-bit A/B lane layout
        FragU afrag[4], bfrag[2];
        #pragma unroll
        for (int mt = 0; mt < 4; ++mt) {
            int base = (wm * 64 + mt * 16 + ln16) * LDSK + g * 8;   // K 0-7/16-23 (+8 for g=1)
            afrag[mt].u[0] = *(const u32x4*)(ab + base);
            afrag[mt].u[1] = *(const u32x4*)(ab + base + 16);
        }
        #pragma unroll
        for (int nt = 0; nt < 2; ++nt) {
            int base = (wn * 32 + nt * 16 + ln16) * LDSK + g * 8;   // K 0-7/16-23 (+8 for g=1)
            bfrag[nt].u[0] = *(const u32x4*)(bb + base);
            bfrag[nt].u[1] = *(const u32x4*)(bb + base + 16);
        }
        #pragma unroll
        for (int mt = 0; mt < 4; ++mt)
            #pragma unroll
            for (int nt = 0; nt < 2; ++nt)
                acc[mt][nt] = __builtin_amdgcn_wmma_f32_16x16x32_f16(
                    false, afrag[mt].h, false, bfrag[nt].h,
                    (short)0, acc[mt][nt], false, false);

        if (more) storeStage((ks + 1) & 1);          // write OTHER buffer: no race with readers
    }

    // ---- reduce sum-of-squares -> reciprocal norms (fused: no second pass over W)
    #pragma unroll
    for (int j = 0; j < 8; ++j)
        rowpart[(a_row + 16 * j) * 16 + (t & 15)] = asum[j];
    colpart[(t >> 7) * BN + b_c] = bsum;
    __syncthreads();

    if (t < BM) {
        float s = 0.f;
        #pragma unroll
        for (int i = 0; i < 16; ++i) s += rowpart[t * 16 + i];
        rfArr[t] = rsqrtf(s + EPS_);
    } else {
        int c = t - BM;
        rwArr[c] = rsqrtf(colpart[c] + colpart[BN + c] + EPS_);
    }
    __syncthreads();

    // ---- epilogue: normalize, clip, per-sample margin, x5, non-temporal store
    #pragma unroll
    for (int mt = 0; mt < 4; ++mt) {
        #pragma unroll
        for (int nt = 0; nt < 2; ++nt) {
            int cl = wn * 32 + nt * 16 + ln16;
            int cg = c0 + cl;
            if (cg >= Cn) continue;
            float rwv = rwArr[cl];
            v8f a = acc[mt][nt];
            #pragma unroll
            for (int v = 0; v < 8; ++v) {
                int ml = wm * 64 + mt * 16 + v + 8 * g;   // C/D layout: VGPR v -> M = v + 8*(lane/16)
                int mg = m0 + ml;
                if (mg >= Bn) continue;
                float s = a[v] * rfArr[ml] * rwv;
                s = fminf(1.f, fmaxf(-1.f, s));
                if (cg == labArr[ml]) s -= margArr[ml];
                __builtin_nontemporal_store(SCALE_ * s, out + (size_t)mg * Cn + cg);
            }
        }
    }
}

extern "C" void kernel_launch(void* const* d_in, const int* in_sizes, int n_in,
                              void* d_out, int out_size, void* d_ws, size_t ws_size,
                              hipStream_t stream) {
    (void)n_in; (void)out_size; (void)d_ws; (void)ws_size;
    const float* feats  = (const float*)d_in[0];
    const int*   labels = (const int*)d_in[1];
    const float* weight = (const float*)d_in[2];
    float*       out    = (float*)d_out;

    const int Bn = in_sizes[1];                       // labels: [B]
    const int Dn = (Bn > 0) ? in_sizes[0] / Bn : 0;   // feats:  [B,D]
    const int Cn = (Dn > 0) ? in_sizes[2] / Dn : 0;   // weight: [D,C]

    dim3 grid((Bn + BM - 1) / BM, (Cn + BN - 1) / BN);  // x = M (4): W-sharing blocks adjacent
    adm_softmax_wmma_kernel<<<grid, 256, 0, stream>>>(feats, labels, weight, out, Bn, Dn, Cn);
}
